// PolicyNet_35673998360889
// MI455X (gfx1250) — matrix operand order repaired and checked by reading, not compiled, not run on hardware
//
#include <hip/hip_runtime.h>
#include <hip/hip_bf16.h>

typedef __attribute__((ext_vector_type(16))) _Float16 v16h;
typedef __attribute__((ext_vector_type(8)))  _Float16 v8h;
typedef __attribute__((ext_vector_type(8)))  float    v8f;

#define BSZ        16384
#define HHID       24
#define NOUT       24
#define NHID       16
#define NGATE      96          // 4*H
#define STEPS      47
#define KPAD       64          // padded K (a=0..23, h=24..47, zero=48..63)
#define WAVES      4           // waves per block
#define ROWS_PER_WAVE 16
#define ROWS_PER_BLOCK (WAVES * ROWS_PER_WAVE)

// Hardware tanh (TRANS32 on CDNA5) with branchless fallback
__device__ __forceinline__ float tanh_hw(float x) {
#if __has_builtin(__builtin_amdgcn_tanhf)
    return __builtin_amdgcn_tanhf(x);
#elif __has_builtin(__builtin_amdgcn_tanh_f32)
    return __builtin_amdgcn_tanh_f32(x);
#else
    float e = __expf(2.0f * x);
    return (e - 1.0f) * __builtin_amdgcn_rcpf(e + 1.0f);
#endif
}
// sigmoid via tanh: 1 trans op instead of exp+rcp
__device__ __forceinline__ float sigf(float x) {
    return fmaf(0.5f, tanh_hw(0.5f * x), 0.5f);
}

__device__ __forceinline__ v8f wmma_f16(v16h a, v16h b, v8f c) {
    return __builtin_amdgcn_wmma_f32_16x16x32_f16(
        /*neg_a=*/false, a, /*neg_b=*/false, b,
        /*c_mod=*/(short)0, c, /*reuse_a=*/false, /*reuse_b=*/false);
}

__global__ __launch_bounds__(128) void policy_lstm_kernel(
    const float* __restrict__ W_ih,   // (96,24)
    const float* __restrict__ W_hh,   // (96,24)
    const float* __restrict__ b_ih,   // (96)
    const float* __restrict__ b_hh,   // (96)
    const float* __restrict__ Wl,     // (24,24)
    const float* __restrict__ bl,     // (24)
    const float* __restrict__ rnd,    // (47,16384)
    float* __restrict__ out)          // prob (B,47) ++ actions (B,47)
{
    __shared__ _Float16 Wc[KPAD][NGATE];        // combined gate weights, f16
    __shared__ _Float16 WlP[KPAD][32];          // Wl.T placed at K=24..47
    __shared__ float    bsum[NGATE];            // b_ih + b_hh
    __shared__ float    blS[NOUT];
    __shared__ __align__(16) _Float16 Asta[WAVES][ROWS_PER_WAVE][KPAD];
    __shared__ float    Gates[WAVES][ROWS_PER_WAVE][NGATE];
    __shared__ float    Logits[WAVES][ROWS_PER_WAVE][32];
    // logits B fragments, per-lane layout: [kt][t][lane][16 halves]
    __shared__ __align__(32) _Float16 bqL[2][2][32][16];

    const int tid  = threadIdx.x;
    const int lane = tid & 31;
    const int w    = tid >> 5;

    // ---------------- init phase 1: weights -> LDS (f16), zero staging ----------------
    for (int idx = tid; idx < KPAD * NGATE; idx += 128) {
        int k = idx / NGATE, n = idx % NGATE;
        float v = 0.0f;
        if (k < 24)      v = W_ih[n * 24 + k];
        else if (k < 48) v = W_hh[n * 24 + (k - 24)];
        Wc[k][n] = (_Float16)v;
    }
    for (int idx = tid; idx < KPAD * 32; idx += 128) {
        int k = idx / 32, n = idx % 32;
        float v = 0.0f;
        if (k >= 24 && k < 48 && n < 24) v = Wl[n * 24 + (k - 24)];
        WlP[k][n] = (_Float16)v;
    }
    for (int g = tid; g < NGATE; g += 128) bsum[g] = b_ih[g] + b_hh[g];
    for (int n = tid; n < NOUT; n += 128) blS[n] = bl[n];
    for (int idx = tid; idx < WAVES * ROWS_PER_WAVE * KPAD; idx += 128)
        (&Asta[0][0][0])[idx] = (_Float16)0;
    __syncthreads();

    // ---------------- init phase 2: build fragment-ordered B data ----------------
    // B layout (16-bit, KxN=32x16): lanes 0-15 cover K=0..15, lanes 16-31 K=16..31;
    // element e of v16h holds K = kb + e for column N = 16*t + (lane&15).
    for (int idx = tid; idx < 2 * 2 * 32 * 16; idx += 128) {
        int e2 = idx & 15;
        int ln = (idx >> 4) & 31;
        int t  = (idx >> 9) & 1;
        int kt = (idx >> 10) & 1;
        int nl = ln & 15, kb = (ln >> 4) * 16;
        bqL[kt][t][ln][e2] = WlP[kt * 32 + kb + e2][t * 16 + nl];
    }

    const int nloc  = lane & 15;
    const int khalf = lane >> 4;      // 0 or 1
    const int kbB   = khalf * 16;

    v16h bg[2][6];                    // gate weights (register resident): [k-tile][n-tile]
#pragma unroll
    for (int kt = 0; kt < 2; ++kt)
#pragma unroll
        for (int t = 0; t < 6; ++t) {
            v16h f;
#pragma unroll
            for (int e = 0; e < 16; ++e)
                f[e] = Wc[kt * 32 + kbB + e][t * 16 + nloc];
            bg[kt][t] = f;
        }
    __syncthreads();

    // A layout (16-bit, MxK=16x32): lane m=lane&15; 16 contiguous K
    // at K = (lane>>4)*16 .. +15 (matches B fragment order).
    const int mA  = lane & 15;
    const int kbA = khalf * 16;
    const _Float16* arow = &Asta[w][mA][0];

    auto loadA = [&](int kTile) -> v16h {
        v8h lo = *(const v8h*)(arow + kTile * 32 + kbA);
        v8h hi = *(const v8h*)(arow + kTile * 32 + kbA + 8);
        return __builtin_shufflevector(lo, hi,
            0, 1, 2, 3, 4, 5, 6, 7, 8, 9, 10, 11, 12, 13, 14, 15);
    };
    auto loadBq = [&](int kt, int t) -> v16h {
        return *(const v16h*)&bqL[kt][t][lane][0];   // 2x ds_load_b128
    };

    // per-lane persistent cell state: element e = lane + 32*q -> (row=e/24, j=e%24)
    float cst[12];
#pragma unroll
    for (int q = 0; q < 12; ++q) cst[q] = 0.0f;
    int prevAct = -1;

    const int mBase = khalf * 8;      // C/D layout: row = vgpr + 8*(lane>=16)
    const int rowg0 = blockIdx.x * ROWS_PER_BLOCK + w * ROWS_PER_WAVE;

    // warm the uniforms for step 0
    if (lane < 16) __builtin_prefetch(&rnd[rowg0 + lane], 0, 3);

    // ---------------- recurrence ----------------
#pragma unroll 1
    for (int i = 0; i < STEPS; ++i) {
        // gates(16x96) = A(16x64) @ Wc(64x96) : 12 WMMAs
        v16h a0 = loadA(0);
        v16h a1 = loadA(1);
#pragma unroll
        for (int t = 0; t < 6; ++t) {
            v8f acc = {};
            acc = wmma_f16(a0, bg[0][t], acc);
            acc = wmma_f16(a1, bg[1][t], acc);
#pragma unroll
            for (int r = 0; r < 8; ++r)
                Gates[w][mBase + r][t * 16 + nloc] = acc[r];
        }

        // LSTM elementwise; c stays in registers, h -> A staging (K = 24+j)
#pragma unroll
        for (int q = 0; q < 12; ++q) {
            int e   = lane + 32 * q;
            int row = e / 24;
            int j   = e % 24;
            float ig = Gates[w][row][j]      + bsum[j];
            float fg = Gates[w][row][j + 24] + bsum[j + 24];
            float gg = Gates[w][row][j + 48] + bsum[j + 48];
            float og = Gates[w][row][j + 72] + bsum[j + 72];
            float cv = sigf(fg) * cst[q] + sigf(ig) * tanh_hw(gg);
            cst[q] = cv;
            float hv = sigf(og) * tanh_hw(cv);
            Asta[w][row][24 + j] = (_Float16)hv;
        }

        // logits(16x24) = A(16x64) @ WlP(64x32) : 4 WMMAs
        // (a-region of A multiplies zero rows of WlP, so same staging works)
        v16h h0 = loadA(0);
        v16h h1 = loadA(1);
#pragma unroll
        for (int t = 0; t < 2; ++t) {
            v8f acc = {};
            acc = wmma_f16(h0, loadBq(0, t), acc);
            acc = wmma_f16(h1, loadBq(1, t), acc);
#pragma unroll
            for (int r = 0; r < 8; ++r)
                Logits[w][mBase + r][t * 16 + nloc] = acc[r];
        }

        // masked softmax + inverse-CDF sampling: lane L<16 owns row L.
        // Logits are bounded (|h|<1, small weights) so exp() needs no max-sub;
        // compare unnormalized cumsum against r*sum (identical result).
        if (lane < 16) {
            int row  = lane;
            int rowg = rowg0 + row;
            float r  = rnd[i * BSZ + rowg];
            if (i + 1 < STEPS)   // hide next step's global latency
                __builtin_prefetch(&rnd[(i + 1) * BSZ + rowg], 0, 3);

            // mask = [lo, hi): odd step -> [0, counter), even step -> [16, 24)
            bool odd = (i & 1) != 0;
            int lo = odd ? 0 : NHID;
            int hi = odd ? ((i + 1) >> 1) : NOUT;

            float ex[24];
            float sum = 0.0f;
#pragma unroll
            for (int col = 0; col < NOUT; ++col) {
                bool m = (col >= lo) && (col < hi);
                float v = Logits[w][row][col] + blS[col];
                float e = m ? __expf(v) : 0.0f;
                ex[col] = e;
                sum += e;
            }
            float thresh = r * sum;

            float cum = 0.0f, pe = 0.0f;
            int act = 0;
            bool found = false;
#pragma unroll
            for (int col = 0; col < NOUT; ++col) {
                cum += ex[col];
                if (!found && cum > thresh) { act = col; pe = ex[col]; found = true; }
            }
            float inv = __builtin_amdgcn_rcpf(sum);
            float pr = (found ? pe : ex[0]) * inv;   // !found mimics argmax-of-all-false
            if (!found) act = 0;

            out[rowg * STEPS + i] = pr;
            out[BSZ * STEPS + rowg * STEPS + i] = (float)act;

            // maintain one-hot a in A staging (K = 0..23)
            if (prevAct >= 0) Asta[w][row][prevAct] = (_Float16)0;
            Asta[w][row][act] = (_Float16)1;
            prevAct = act;
        }
    }
}

extern "C" void kernel_launch(void* const* d_in, const int* in_sizes, int n_in,
                              void* d_out, int out_size, void* d_ws, size_t ws_size,
                              hipStream_t stream) {
    (void)in_sizes; (void)n_in; (void)out_size; (void)d_ws; (void)ws_size;
    const float* W_ih = (const float*)d_in[0];
    const float* W_hh = (const float*)d_in[1];
    const float* b_ih = (const float*)d_in[2];
    const float* b_hh = (const float*)d_in[3];
    const float* Wl   = (const float*)d_in[4];
    const float* bl   = (const float*)d_in[5];
    const float* rnd  = (const float*)d_in[6];
    float* out = (float*)d_out;

    dim3 grid(BSZ / ROWS_PER_BLOCK);   // 256 blocks
    dim3 block(128);                   // 4 wave32s
    policy_lstm_kernel<<<grid, block, 0, stream>>>(
        W_ih, W_hh, b_ih, b_hh, Wl, bl, rnd, out);
}
